// SinkhornScorer_39075612459657
// MI455X (gfx1250) — matrix operand order repaired and checked by reading, not compiled, not run on hardware
//
#include <hip/hip_runtime.h>
#include <hip/hip_bf16.h>

typedef _Float16 v16h __attribute__((ext_vector_type(16)));
typedef float    v8f  __attribute__((ext_vector_type(8)));

#define B_TOT   4096
#define NXR     64
#define HD      256
#define ITERS   20
#define CS      67                      // coupling row stride (floats); gcd(67,64)=1 -> conflict-free cols
#define NEG_INF (-3.0e38f)
#define NORM_C  (-4.852030263919617f)   // -log(128)
#define LOGBIN  (-0.6931471805599453f)  // log(64) - log(128)
#define INV_REG 10.0f
#define REG_F   0.1f
#define TEMP_F  0.1f

__device__ __forceinline__ float wred_max(float v) {
#pragma unroll
  for (int o = 16; o > 0; o >>= 1) v = fmaxf(v, __shfl_xor(v, o, 32));
  return v;
}
__device__ __forceinline__ float wred_sum(float v) {
#pragma unroll
  for (int o = 16; o > 0; o >>= 1) v += __shfl_xor(v, o, 32);
  return v;
}

__global__ __launch_bounds__(256)
void sinkhorn_fused_kernel(const float* __restrict__ x, const float* __restrict__ y,
                           const float* __restrict__ gam, const float* __restrict__ bet,
                           const float* __restrict__ wdb, const float* __restrict__ bdb,
                           float* __restrict__ out_m, float* __restrict__ out_s)
{
  // 64KB main region: xn(32KB f16) | yn(32KB f16); couplings alias xn after GEMM.
  __shared__ __align__(16) unsigned char smem[65536];
  __shared__ float s_alpha[128];
  __shared__ float s_u[66], s_v[66], s_red[8];

  _Float16* xn = (_Float16*)smem;
  _Float16* yn = (_Float16*)(smem + 32768);
  float*    Cl = (float*)smem;                 // valid only after GEMM barrier

  const int tid  = threadIdx.x;
  const int lane = tid & 31;
  const int wv   = tid >> 5;
  const int b    = blockIdx.x;

  // ---- per-lane LayerNorm->Linear constants (each wave redundantly covers all 256 h) ----
  float gw[8]; float sgw_l = 0.f, bw_l = 0.f;
#pragma unroll
  for (int k = 0; k < 8; ++k) {
    int h = lane + 32 * k;
    float wk = wdb[h];
    gw[k] = gam[h] * wk;
    sgw_l += gw[k];
    bw_l  += bet[h] * wk;
  }
  const float SGW = wred_sum(sgw_l);
  const float BW  = wred_sum(bw_l) + bdb[0];

  // ---- Stage 1: fused l2norm (-> f16 LDS) + dustbin logit per row ----
  const float* srcx = x + (size_t)b * NXR * HD;
  const float* srcy = y + (size_t)b * NXR * HD;
#pragma unroll 1
  for (int rr = wv; rr < 128; rr += 8) {
    const float* src = (rr < 64) ? (srcx + rr * HD) : (srcy + (rr - 64) * HD);
    _Float16*    dst = (rr < 64) ? (xn + rr * HD)   : (yn + (rr - 64) * HD);
    float vbuf[8]; float s1 = 0.f, s2 = 0.f, s3 = 0.f;
#pragma unroll
    for (int k = 0; k < 8; ++k) {
      float t = src[lane + 32 * k];
      vbuf[k] = t; s1 += t; s2 += t * t; s3 += t * gw[k];
    }
    s1 = wred_sum(s1); s2 = wred_sum(s2); s3 = wred_sum(s3);
    float inv_n = 1.0f / fmaxf(sqrtf(s2), 1e-12f);
#pragma unroll
    for (int k = 0; k < 8; ++k) dst[lane + 32 * k] = (_Float16)(vbuf[k] * inv_n);
    if (lane == 0) {
      float mu  = s1 * (1.0f / HD);
      float var = s2 * (1.0f / HD) - mu * mu;       // biased, like torch LayerNorm
      float inv_std = rsqrtf(var + 1e-5f);
      s_alpha[rr] = tanhf(inv_std * (s3 - mu * SGW) + BW);
    }
  }
  __syncthreads();

  // ---- Stage 2: scores = xn * yn^T via v_wmma_f32_16x16x32_f16 ----
  // 16 tiles of 16x16; wave wv owns tiles (ti0,tj) and (ti0+2,tj) sharing the B fragment.
  const int tj   = wv & 3;
  const int ti0  = wv >> 2;
  const int half = lane >> 4;
  const int nib  = lane & 15;
  const _Float16* arow0 = xn + (ti0 * 16 + nib) * HD;
  const _Float16* arow1 = xn + ((ti0 + 2) * 16 + nib) * HD;
  const _Float16* brow  = yn + (tj * 16 + nib) * HD;
  v8f acc0 = {}; v8f acc1 = {};
#pragma unroll
  for (int kb = 0; kb < 8; ++kb) {
    const int kbase = kb * 32;
    union { v16h v; unsigned u[8]; } af0, af1, bf;
#pragma unroll
    for (int q = 0; q < 8; ++q) {
      // A layout (16-bit 16x32): lane half h covers K = h*8 + {0..7} and 16 + h*8 + {0..7}, pairs per VGPR
      int kkA = kbase + ((q >> 2) * 16) + half * 8 + (q & 3) * 2;
      // B layout (16-bit 32x16): same K order as A: lane half h covers K = h*8 + {0..7} and 16 + h*8 + {0..7}
      int kkB = kbase + ((q >> 2) * 16) + half * 8 + (q & 3) * 2;
      af0.u[q] = *(const unsigned*)(arow0 + kkA);
      af1.u[q] = *(const unsigned*)(arow1 + kkA);
      bf.u[q]  = *(const unsigned*)(brow  + kkB);
    }
    acc0 = __builtin_amdgcn_wmma_f32_16x16x32_f16(false, af0.v, false, bf.v, (short)0, acc0, false, false);
    acc1 = __builtin_amdgcn_wmma_f32_16x16x32_f16(false, af1.v, false, bf.v, (short)0, acc1, false, false);
  }
  __syncthreads();   // all xn/yn reads done; safe to overwrite with couplings

  // ---- write couplings (scores / reg) + dustbins + u/v init ----
  {
    const int col = tj * 16 + nib;
#pragma unroll
    for (int r = 0; r < 8; ++r) {
      int row0 = ti0 * 16 + half * 8 + r;
      int row1 = (ti0 + 2) * 16 + half * 8 + r;
      Cl[row0 * CS + col] = acc0[r] * INV_REG;
      Cl[row1 * CS + col] = acc1[r] * INV_REG;
    }
  }
  if (tid < 64) {
    Cl[tid * CS + 64] = s_alpha[tid] * INV_REG;        // alpha_x column
    Cl[64 * CS + tid] = s_alpha[64 + tid] * INV_REG;   // alpha_y row
  }
  if (tid == 64) Cl[64 * CS + 64] = -100.0f * INV_REG; // ALPHA_BOTH / reg
  if (tid < 65) { s_u[tid] = 0.f; s_v[tid] = 0.f; }
  __syncthreads();

  // ---- Stage 3: 20 Sinkhorn iterations, one wave per row/column ----
#pragma unroll 1
  for (int it = 0; it < ITERS; ++it) {
    for (int i = wv; i < 65; i += 8) {                 // u update (row logsumexp)
      const float* Ci = Cl + i * CS;
      float t0 = Ci[lane]      + s_v[lane];
      float t1 = Ci[lane + 32] + s_v[lane + 32];
      float t2 = (lane == 0) ? (Ci[64] + s_v[64]) : NEG_INF;
      float m  = wred_max(fmaxf(fmaxf(t0, t1), t2));
      float s  = __expf(t0 - m) + __expf(t1 - m) + ((lane == 0) ? __expf(t2 - m) : 0.f);
      s = wred_sum(s);
      if (lane == 0) s_u[i] = ((i < 64) ? NORM_C : LOGBIN) - (m + __logf(s));
    }
    __syncthreads();
    for (int j = wv; j < 65; j += 8) {                 // v update (col logsumexp)
      float t0 = Cl[lane * CS + j]        + s_u[lane];
      float t1 = Cl[(lane + 32) * CS + j] + s_u[lane + 32];
      float t2 = (lane == 0) ? (Cl[64 * CS + j] + s_u[64]) : NEG_INF;
      float m  = wred_max(fmaxf(fmaxf(t0, t1), t2));
      float s  = __expf(t0 - m) + __expf(t1 - m) + ((lane == 0) ? __expf(t2 - m) : 0.f);
      s = wred_sum(s);
      if (lane == 0) s_v[j] = ((j < 64) ? NORM_C : LOGBIN) - (m + __logf(s));
    }
    __syncthreads();
  }

  // ---- Stage 4: matching output + score reduction ----
  float* om = out_m + (size_t)b * (65 * 65);
  float local = 0.f;
#pragma unroll 1
  for (int idx = tid; idx < 65 * 65; idx += 256) {
    int i = idx / 65;
    int j = idx - i * 65;
    float c   = Cl[i * CS + j];
    float val = c + s_u[i] + s_v[j] - NORM_C;          // matching = couplings + u + v - norm
    om[idx] = val;
    if ((i < 64) && (j < 64)) local += __expf(val) * c; // scores = couplings * REG, folded below
  }
  local = wred_sum(local);
  if (lane == 0) s_red[wv] = local;
  __syncthreads();
  if (tid == 0) {
    float t = 0.f;
#pragma unroll
    for (int k = 0; k < 8; ++k) t += s_red[k];
    out_s[b] = t * (REG_F / TEMP_F);
  }
}

extern "C" void kernel_launch(void* const* d_in, const int* in_sizes, int n_in,
                              void* d_out, int out_size, void* d_ws, size_t ws_size,
                              hipStream_t stream) {
  const float* x   = (const float*)d_in[0];
  const float* y   = (const float*)d_in[1];
  const float* gam = (const float*)d_in[2];
  const float* bet = (const float*)d_in[3];
  const float* wdb = (const float*)d_in[4];
  const float* bdb = (const float*)d_in[5];
  float* out_m = (float*)d_out;
  float* out_s = out_m + (size_t)B_TOT * 65 * 65;
  hipLaunchKernelGGL(sinkhorn_fused_kernel, dim3(B_TOT), dim3(256), 0, stream,
                     x, y, gam, bet, wdb, bdb, out_m, out_s);
}
